// VirusHostClassifier_8804682957194
// MI455X (gfx1250) — compile-verified, matrix operand order checked by reading, not run on hardware
//
#include <hip/hip_runtime.h>
#include <hip/hip_bf16.h>
#include <stdint.h>

#define B_    32
#define C_    32
#define S_    16
#define E_    1280
#define R_    256
#define H_    4
#define HD_   64
#define FF_   512
#define HOSTD 64
#define VIRD  32
#define METAD 3
#define HEADIN 8291
#define HC1   128

typedef __attribute__((ext_vector_type(16))) _Float16 v16h;
typedef __attribute__((ext_vector_type(8)))  float    v8f;

// ---------------------------------------------------------------------------
// f32 -> f16 bulk convert (weights, once per launch; cost is negligible vs the
// 84 MB embs stream).
// ---------------------------------------------------------------------------
__global__ void cvt_f16_kernel(const float* __restrict__ in,
                               _Float16* __restrict__ out, int n)
{
    const int i4 = (blockIdx.x * blockDim.x + threadIdx.x) * 4;
    if (i4 + 3 < n) {
        const float4 f = *(const float4*)(in + i4);
        out[i4 + 0] = (_Float16)f.x;
        out[i4 + 1] = (_Float16)f.y;
        out[i4 + 2] = (_Float16)f.z;
        out[i4 + 3] = (_Float16)f.w;
    } else {
        for (int i = i4; i < n; ++i) out[i] = (_Float16)in[i];
    }
}

// ---------------------------------------------------------------------------
// Stage 1: masked softmax over S=16 slots per (b,c) segment, weighted sum over
// embs -> agg_h[B*C, E] in f16 (feeds WMMA GEMM1 directly).  Owns the dominant
// 84 MB HBM read; streams embs once, coalesced across E.
// ---------------------------------------------------------------------------
__global__ void agg_kernel(const float* __restrict__ embs,
                           const int* __restrict__ indices,
                           const unsigned char* __restrict__ mask,
                           const float* __restrict__ fw,
                           _Float16* __restrict__ agg_h)
{
    __shared__ float attn[S_];
    const int seg = blockIdx.x;            // b*C + c
    const int tid = threadIdx.x;
    if (tid == 0) {
        float w[S_];
        bool any = false;
        float mx = -3.4e38f;
        for (int s = 0; s < S_; ++s) {
            const bool mk = mask[seg * S_ + s] != 0;
            any |= mk;
            const float ww = mk ? fw[indices[seg * S_ + s]] : -1e30f;
            w[s] = ww;
            mx = fmaxf(mx, ww);
        }
        float sum = 0.f;
        for (int s = 0; s < S_; ++s) { w[s] = __expf(w[s] - mx); sum += w[s]; }
        const float inv = any ? (1.0f / sum) : 0.0f;   // empty segment -> zeros
        for (int s = 0; s < S_; ++s) attn[s] = w[s] * inv;
    }
    __syncthreads();
    const float* base = embs + (size_t)seg * S_ * E_;
    for (int e = tid; e < E_; e += blockDim.x) {
        __builtin_prefetch(base + e + 256, 0, 0);      // global_prefetch_b8
        float acc = 0.f;
#pragma unroll
        for (int s = 0; s < S_; ++s)
            acc += attn[s] * base[s * E_ + e];
        agg_h[(size_t)seg * E_ + e] = (_Float16)acc;
    }
}

// ---------------------------------------------------------------------------
// WMMA GEMM on f16 operands:
//   C[M,N] = act( A[M,K] @ W[N,K]^T + bias ),  f32 accumulate.
// One wave computes a 16x64 output strip (4 accumulators): the A fragment is
// loaded once per K-step (2x GLOBAL_LOAD_B128) and reused by 4 WMMAs.
// Fragment layout per CDNA5 ISA 7.12.2: lane (hi,lo) holds
// K = 8*hi + [0..7] and 16 + 8*hi + [0..7] of row lo -> two contiguous
// 16-byte runs in row-major f16.  M mult 16, N mult 64, K mult 32.
// ---------------------------------------------------------------------------
__global__ void wmma_gemm_h(const _Float16* __restrict__ Ah,   // [M,K]
                            const _Float16* __restrict__ Wh,   // [N,K]
                            const float* __restrict__ bias,
                            float* __restrict__ Cout,          // nullable
                            _Float16* __restrict__ Couth,      // nullable
                            int M, int N, int K, int act)
{
    const int wave   = (int)((blockIdx.x * blockDim.x + threadIdx.x) >> 5);
    const int lane   = (int)(threadIdx.x & 31);
    const int tilesN = N >> 6;                // 64-wide strips
    const int tm     = wave / tilesN;
    const int tg     = wave - tm * tilesN;
    if (tm * 16 >= M) return;                 // whole wave exits; EXEC stays full

    const int lo = lane & 15;
    const int hi = lane >> 4;
    const _Float16* ap  = Ah + (size_t)(tm * 16 + lo) * K + 8 * hi;
    const _Float16* wp0 = Wh + (size_t)(tg * 64 + lo) * K + 8 * hi;
    const size_t wstep = (size_t)16 * K;

    v8f acc[4] = {};
    for (int k0 = 0; k0 < K; k0 += 32) {
        union { v16h h; uint4 q[2]; } a;
        a.q[0] = *(const uint4*)(ap + k0);
        a.q[1] = *(const uint4*)(ap + k0 + 16);
#pragma unroll
        for (int j = 0; j < 4; ++j) {
            union { v16h h; uint4 q[2]; } b;
            const _Float16* wp = wp0 + (size_t)j * wstep;
            b.q[0] = *(const uint4*)(wp + k0);
            b.q[1] = *(const uint4*)(wp + k0 + 16);
            // (neg_a, A, neg_b, B, c_mod, C, reuse_a, reuse_b)
            acc[j] = __builtin_amdgcn_wmma_f32_16x16x32_f16(
                false, a.h, false, b.h, (short)0, acc[j], false, false);
        }
    }

#pragma unroll
    for (int j = 0; j < 4; ++j) {
        const int   ncol = tg * 64 + j * 16 + lo;
        const float bv   = bias ? bias[ncol] : 0.0f;
#pragma unroll
        for (int v = 0; v < 8; ++v) {
            const int m = tm * 16 + hi * 8 + v;   // D: VGPR v -> M = v (+8 hi half)
            float val = acc[j][v] + bv;
            if (act) val = fmaxf(val, 0.0f);
            if (Cout)  Cout [(size_t)m * N + ncol] = val;
            if (Couth) Couth[(size_t)m * N + ncol] = (_Float16)val;
        }
    }
}

// ---------------------------------------------------------------------------
// Attention core: per (b, head) block. 32x32 scores, softmax over k, A@V.
// Output written in f16 (feeds the Wo WMMA GEMM).  LDS: 28 KB.
// ---------------------------------------------------------------------------
__global__ void attn_kernel(const float* __restrict__ qkv,
                            _Float16* __restrict__ oh)
{
    __shared__ float qs[C_][HD_];
    __shared__ float ks[C_][HD_];
    __shared__ float vs[C_][HD_];
    __shared__ float sc[C_][C_];
    const int bh = blockIdx.x;
    const int b = bh / H_, h = bh % H_;
    const int tid = threadIdx.x;

    for (int i = tid; i < C_ * HD_; i += 256) {
        const int c = i / HD_, d = i % HD_;
        const size_t base = (size_t)(b * C_ + c) * (3 * R_) + h * HD_ + d;
        qs[c][d] = qkv[base];
        ks[c][d] = qkv[base + R_];
        vs[c][d] = qkv[base + 2 * R_];
    }
    __syncthreads();

    for (int i = tid; i < C_ * C_; i += 256) {
        const int r = i / C_, c = i % C_;
        float acc = 0.f;
#pragma unroll 8
        for (int d = 0; d < HD_; ++d) acc += qs[r][d] * ks[c][d];
        sc[r][c] = acc * 0.125f;              // 1/sqrt(64)
    }
    __syncthreads();

    if (tid < C_) {
        float mx = -3.4e38f;
        for (int c = 0; c < C_; ++c) mx = fmaxf(mx, sc[tid][c]);
        float sum = 0.f;
        for (int c = 0; c < C_; ++c) { const float e = __expf(sc[tid][c] - mx); sc[tid][c] = e; sum += e; }
        const float inv = 1.f / sum;
        for (int c = 0; c < C_; ++c) sc[tid][c] *= inv;
    }
    __syncthreads();

    for (int i = tid; i < C_ * HD_; i += 256) {
        const int r = i / HD_, d = i % HD_;
        float acc = 0.f;
#pragma unroll 8
        for (int k = 0; k < C_; ++k) acc += sc[r][k] * vs[k][d];
        oh[(size_t)(b * C_ + r) * R_ + h * HD_ + d] = (_Float16)acc;
    }
}

// ---------------------------------------------------------------------------
// xout = LayerNorm(xin + res) * g + b  (+ optional f16 copy for the next GEMM)
// ---------------------------------------------------------------------------
__global__ void add_ln_kernel(const float* __restrict__ xin,
                              const float* __restrict__ res,
                              const float* __restrict__ g,
                              const float* __restrict__ bb,
                              float* __restrict__ xout,
                              _Float16* __restrict__ xouth)
{
    __shared__ float red[R_];
    const int row = blockIdx.x;
    const int t = threadIdx.x;
    const size_t idx = (size_t)row * R_ + t;
    const float v = xin[idx] + res[idx];
    red[t] = v;
    __syncthreads();
    for (int off = R_ / 2; off > 0; off >>= 1) { if (t < off) red[t] += red[t + off]; __syncthreads(); }
    const float mean = red[0] * (1.0f / R_);
    __syncthreads();
    const float d = v - mean;
    red[t] = d * d;
    __syncthreads();
    for (int off = R_ / 2; off > 0; off >>= 1) { if (t < off) red[t] += red[t + off]; __syncthreads(); }
    const float var = red[0] * (1.0f / R_);
    const float r = d * rsqrtf(var + 1e-5f) * g[t] + bb[t];
    xout[idx] = r;
    if (xouth) xouth[idx] = (_Float16)r;
}

// ---------------------------------------------------------------------------
// Head: feat = [x2(8192) | host(64) | virus(32) | meta(3)];
// h = relu(feat@Wc1^T + bc1); logits = h@Wc2^T + bc2.
// One block per batch, 128 threads (one per hidden unit), LDS tree reduce.
// ---------------------------------------------------------------------------
__global__ void head_kernel(const float* __restrict__ x2,
                            const float* __restrict__ host_cat,
                            const float* __restrict__ virus_cat,
                            const float* __restrict__ extra_meta,
                            const float* __restrict__ Wc1,
                            const float* __restrict__ bc1,
                            const float* __restrict__ Wc2,
                            const float* __restrict__ bc2,
                            float* __restrict__ out)
{
    __shared__ float hbuf[HC1];
    const int b = blockIdx.x;
    const int n = threadIdx.x;                      // 0..127
    const float* wr = Wc1 + (size_t)n * HEADIN;
    const float* xb = x2 + (size_t)b * (C_ * R_);
    float acc = bc1[n];
    for (int j = 0; j < C_ * R_; ++j) acc += xb[j] * wr[j];
    for (int j = 0; j < HOSTD; ++j)  acc += host_cat[b * HOSTD + j]  * wr[C_ * R_ + j];
    for (int j = 0; j < VIRD; ++j)   acc += virus_cat[b * VIRD + j]  * wr[C_ * R_ + HOSTD + j];
    for (int j = 0; j < METAD; ++j)  acc += extra_meta[b * METAD + j] * wr[C_ * R_ + HOSTD + VIRD + j];
    hbuf[n] = fmaxf(acc, 0.f) * Wc2[n];
    __syncthreads();
    for (int off = HC1 / 2; off > 0; off >>= 1) {
        if (n < off) hbuf[n] += hbuf[n + off];
        __syncthreads();
    }
    if (n == 0) out[b] = hbuf[0] + bc2[0];
}

// ---------------------------------------------------------------------------
extern "C" void kernel_launch(void* const* d_in, const int* in_sizes, int n_in,
                              void* d_out, int out_size, void* d_ws, size_t ws_size,
                              hipStream_t stream)
{
    const float* embs       = (const float*)d_in[0];
    const int*   indices    = (const int*)d_in[1];
    const unsigned char* mask = (const unsigned char*)d_in[2];
    const float* host_cat   = (const float*)d_in[3];
    const float* virus_cat  = (const float*)d_in[4];
    const float* extra_meta = (const float*)d_in[5];
    const float* fw         = (const float*)d_in[6];
    const float* Wr   = (const float*)d_in[7];
    const float* br   = (const float*)d_in[8];
    const float* Wqkv = (const float*)d_in[9];
    const float* bqkv = (const float*)d_in[10];
    const float* Wo   = (const float*)d_in[11];
    const float* bo   = (const float*)d_in[12];
    const float* ln1_g = (const float*)d_in[13];
    const float* ln1_b = (const float*)d_in[14];
    const float* W1   = (const float*)d_in[15];
    const float* b1   = (const float*)d_in[16];
    const float* W2   = (const float*)d_in[17];
    const float* b2   = (const float*)d_in[18];
    const float* ln2_g = (const float*)d_in[19];
    const float* ln2_b = (const float*)d_in[20];
    const float* Wc1  = (const float*)d_in[21];
    const float* bc1  = (const float*)d_in[22];
    const float* Wc2  = (const float*)d_in[23];
    const float* bc2  = (const float*)d_in[24];
    float* out = (float*)d_out;

    char* wsb = (char*)d_ws;
    // --- persistent region (byte offsets, 256-aligned) ---
    const size_t X     = 0;          // f32  1024*256   (x after Wr)
    const size_t XH    = 1048576;    // f16  1024*256
    const size_t X1    = 1572864;    // f32  1024*256   (x after ln1)
    const size_t X1H   = 2621440;    // f16  1024*256
    const size_t X2    = 3145728;    // f32  1024*256   (x after ln2)
    const size_t WRH   = 4194304;    // f16  256*1280
    const size_t WQKVH = 4849664;    // f16  768*256
    const size_t WOH   = 5242880;    // f16  256*256
    const size_t W1H   = 5373952;    // f16  512*256
    const size_t W2H   = 5636096;    // f16  256*512
    // --- overlaid scratch region (4.5 MB) ---
    const size_t SCR   = 5898240;
    const size_t AGGH  = SCR;            // f16 1024*1280 (dead after GEMM1)
    const size_t QKV   = SCR;            // f32 1024*768  (dead after attn)
    const size_t OH    = SCR + 3145728;  // f16 1024*256  (dead after GEMM3)
    const size_t O2    = SCR + 3670016;  // f32 1024*256  (dead after ln1)
    const size_t FFHH  = SCR;            // f16 1024*512  (dead after GEMM5)
    const size_t FFO   = SCR + 1048576;  // f32 1024*256  (dead after ln2)
    // total footprint: SCR + 4718592 = 10,616,832 bytes (~10.1 MB)

    const int MC = B_ * C_;              // 1024 rows

    // weights -> f16 (once per launch)
    cvt_f16_kernel<<<(R_ * E_ / 4 + 255) / 256, 256, 0, stream>>>(Wr,   (_Float16*)(wsb + WRH),   R_ * E_);
    cvt_f16_kernel<<<(3 * R_ * R_ / 4 + 255) / 256, 256, 0, stream>>>(Wqkv, (_Float16*)(wsb + WQKVH), 3 * R_ * R_);
    cvt_f16_kernel<<<(R_ * R_ / 4 + 255) / 256, 256, 0, stream>>>(Wo,   (_Float16*)(wsb + WOH),   R_ * R_);
    cvt_f16_kernel<<<(FF_ * R_ / 4 + 255) / 256, 256, 0, stream>>>(W1,  (_Float16*)(wsb + W1H),   FF_ * R_);
    cvt_f16_kernel<<<(R_ * FF_ / 4 + 255) / 256, 256, 0, stream>>>(W2,  (_Float16*)(wsb + W2H),   R_ * FF_);

    agg_kernel<<<MC, 256, 0, stream>>>(embs, indices, mask, fw, (_Float16*)(wsb + AGGH));

    // x = agg @ Wr^T + br              [1024,1280]x[1280,256]
    wmma_gemm_h<<<(MC / 16) * (R_ / 64) / 8, 256, 0, stream>>>(
        (const _Float16*)(wsb + AGGH), (const _Float16*)(wsb + WRH), br,
        (float*)(wsb + X), (_Float16*)(wsb + XH), MC, R_, E_, 0);
    // qkv = x @ Wqkv^T + bqkv          [1024,256]x[256,768]
    wmma_gemm_h<<<(MC / 16) * (3 * R_ / 64) / 8, 256, 0, stream>>>(
        (const _Float16*)(wsb + XH), (const _Float16*)(wsb + WQKVH), bqkv,
        (float*)(wsb + QKV), (_Float16*)nullptr, MC, 3 * R_, R_, 0);

    attn_kernel<<<B_ * H_, 256, 0, stream>>>((const float*)(wsb + QKV), (_Float16*)(wsb + OH));

    // o2 = o @ Wo^T + bo               [1024,256]x[256,256]
    wmma_gemm_h<<<(MC / 16) * (R_ / 64) / 8, 256, 0, stream>>>(
        (const _Float16*)(wsb + OH), (const _Float16*)(wsb + WOH), bo,
        (float*)(wsb + O2), (_Float16*)nullptr, MC, R_, R_, 0);
    add_ln_kernel<<<MC, R_, 0, stream>>>((const float*)(wsb + X), (const float*)(wsb + O2),
                                         ln1_g, ln1_b,
                                         (float*)(wsb + X1), (_Float16*)(wsb + X1H));

    // ffh = relu(x1 @ W1^T + b1)       [1024,256]x[256,512] -> f16 only
    wmma_gemm_h<<<(MC / 16) * (FF_ / 64) / 8, 256, 0, stream>>>(
        (const _Float16*)(wsb + X1H), (const _Float16*)(wsb + W1H), b1,
        (float*)nullptr, (_Float16*)(wsb + FFHH), MC, FF_, R_, 1);
    // ffo = ffh @ W2^T + b2            [1024,512]x[512,256]
    wmma_gemm_h<<<(MC / 16) * (R_ / 64) / 8, 256, 0, stream>>>(
        (const _Float16*)(wsb + FFHH), (const _Float16*)(wsb + W2H), b2,
        (float*)(wsb + FFO), (_Float16*)nullptr, MC, R_, FF_, 0);
    add_ln_kernel<<<MC, R_, 0, stream>>>((const float*)(wsb + X1), (const float*)(wsb + FFO),
                                         ln2_g, ln2_b,
                                         (float*)(wsb + X2), (_Float16*)nullptr);

    head_kernel<<<B_, HC1, 0, stream>>>((const float*)(wsb + X2), host_cat, virus_cat, extra_meta,
                                        Wc1, bc1, Wc2, bc2, out);
}
